// MultiHeadSelfAttention_89859305766963
// MI455X (gfx1250) — matrix operand order repaired and checked by reading, not compiled, not run on hardware
//
#include <hip/hip_runtime.h>

// ---------------------------------------------------------------------------
// MI455X (gfx1250, wave32) fused multi-head self-attention with RoPE + ALiBi
//   B=2, T=2048, D=1024, H=16, Dh=64
// All GEMMs use v_wmma_f32_16x16x32_bf16 (f32 accumulate).
// Projection GEMMs: 32x64 tile per wave (2 A-frags share each B-frag).
// Attention: flash, with double-buffered global_load_async_to_lds_b128
// staging of K/V tiles + s_wait_asynccnt (ASYNCcnt pipeline).
// ---------------------------------------------------------------------------

#define TOKENS 4096          // B*T
#define DM     1024
#define NH     16
#define DH     64
#define SEQ    2048

typedef __attribute__((ext_vector_type(16))) __bf16 bf16x16;
typedef __attribute__((ext_vector_type(4)))  __bf16 bf16x4;
typedef __attribute__((ext_vector_type(8)))  float  v8f;
typedef int v4i __attribute__((vector_size(16)));

#if __has_builtin(__builtin_amdgcn_global_load_async_to_lds_b128)
#define HAVE_ASYNC_LDS 1
#endif

union FragB {
    bf16x16 v;
    uint4   q[2];
};

__device__ __forceinline__ v8f wmma_bf16(bf16x16 a, bf16x16 b, v8f c) {
    return __builtin_amdgcn_wmma_f32_16x16x32_bf16(
        /*neg_a=*/false, a, /*neg_b=*/false, b,
        /*c_mod=*/(short)0, c, /*reuse_a=*/false, /*reuse_b=*/false);
}

// A fragment (16x32 bf16): p = row_base + k0 + half*16  (row = m0 + (lane&15))
__device__ __forceinline__ bf16x16 load_a_frag(const __bf16* p) {
    FragB f;
    f.q[0] = *(const uint4*)(p);
    f.q[1] = *(const uint4*)(p + 8);
    return f.v;
}
// B fragment (32x16 bf16): p = row_base + k0 + half*16  (row = n0 + (lane&15))
__device__ __forceinline__ bf16x16 load_b_frag(const __bf16* p) {
    FragB f;
    f.q[0] = *(const uint4*)(p);
    f.q[1] = *(const uint4*)(p + 8);
    return f.v;
}

#ifdef HAVE_ASYNC_LDS
// 16 B per lane, global -> LDS via the CDNA5 async engine (ASYNCcnt).
// Builtin signature (from diagnostics): (int4 AS1*, int4 AS3*, imm, imm).
__device__ __forceinline__ void async_cp16(const void* g, void* l) {
    __builtin_amdgcn_global_load_async_to_lds_b128(
        (__attribute__((address_space(1))) v4i*)(v4i*)(void*)g,
        (__attribute__((address_space(3))) v4i*)(v4i*)l,
        /*imm offset=*/0, /*cpol=*/0);
}
#endif

// ---------------------------------------------------------------------------
// Kernel 1: f32 -> bf16 conversion of x and the four weight matrices.
// 8M elements, 4 per thread.
// ---------------------------------------------------------------------------
__global__ __launch_bounds__(256) void convert_kernel(
    const float* __restrict__ x,
    const float* __restrict__ wq, const float* __restrict__ wk,
    const float* __restrict__ wv, const float* __restrict__ wo,
    __bf16* __restrict__ xb, __bf16* __restrict__ wb)
{
    long long gid = (long long)blockIdx.x * 256 + threadIdx.x;   // < 2M
    long long i4  = gid * 4;
    const float* src;
    __bf16* dst;
    long long off;
    if (i4 < 4194304LL) {                       // x: 4096*1024
        src = x; dst = xb; off = i4;
    } else {
        long long w = i4 - 4194304LL;
        int seg = (int)(w >> 20);               // 0..3 : Wq,Wk,Wv,Wo
        off = w & 1048575LL;
        src = (seg == 0) ? wq : (seg == 1) ? wk : (seg == 2) ? wv : wo;
        dst = wb + (long long)seg * 1048576LL;
    }
    float4 v = *(const float4*)(src + off);
    bf16x4 o = { (__bf16)v.x, (__bf16)v.y, (__bf16)v.z, (__bf16)v.w };
    *(bf16x4*)(dst + off) = o;
}

// ---------------------------------------------------------------------------
// Kernel 2: fused QKV projection.  qkv[sel] = xb @ W[sel]^T + b[sel]  (bf16 out)
// One wave computes a 32x64 output tile; 8 waves / block.
// ---------------------------------------------------------------------------
__global__ __launch_bounds__(256) void gemm_qkv_kernel(
    const __bf16* __restrict__ xb, const __bf16* __restrict__ wb,
    const float* __restrict__ bq, const float* __restrict__ bk,
    const float* __restrict__ bv, __bf16* __restrict__ qkv)
{
    const int lane = threadIdx.x & 31;
    const int wave = threadIdx.x >> 5;
    const int hs   = lane >> 4;       // half-wave select
    const int r    = lane & 15;
    int gw = blockIdx.x * 8 + wave;
    const int nStrips = 3072 / 64;    // 48
    int mt = gw / nStrips;            // 0..127
    int ns = gw - mt * nStrips;
    int m0 = mt * 32, n0 = ns * 64;

    v8f acc[2][4] = {};
    const __bf16* pa0 = xb + (size_t)(m0 + r) * DM + hs * 16;
    const __bf16* pa1 = pa0 + (size_t)16 * DM;
    const __bf16* pw  = wb + hs * 16;          // rows 0..3071 -> Wq|Wk|Wv
    for (int k0 = 0; k0 < DM; k0 += 32) {
        __builtin_prefetch(pa0 + k0 + 256, 0, 1);   // global_prefetch_b8
        __builtin_prefetch(pa1 + k0 + 256, 0, 1);
        bf16x16 a0 = load_a_frag(pa0 + k0);
        bf16x16 a1 = load_a_frag(pa1 + k0);
#pragma unroll
        for (int j = 0; j < 4; ++j) {
            bf16x16 b = load_b_frag(pw + (size_t)(n0 + j * 16 + r) * DM + k0);
            acc[0][j] = wmma_bf16(a0, b, acc[0][j]);
            acc[1][j] = wmma_bf16(a1, b, acc[1][j]);
        }
    }
#pragma unroll
    for (int j = 0; j < 4; ++j) {
        int ng  = n0 + j * 16;
        int sel = ng >> 10;
        int nl  = (ng & 1023) + r;
        const float* bias = (sel == 0) ? bq : ((sel == 1) ? bk : bv);
        float bb = bias[nl];
        __bf16* po = qkv + (size_t)sel * 4194304 + nl;
#pragma unroll
        for (int g = 0; g < 2; ++g) {
#pragma unroll
            for (int i = 0; i < 8; ++i) {
                int t = m0 + g * 16 + hs * 8 + i;
                po[(size_t)t * DM] = (__bf16)(acc[g][j][i] + bb);
            }
        }
    }
}

// ---------------------------------------------------------------------------
// Kernel 3: RoPE on q,k + relayout to head-major [b,h,t,d]; V transposed to
// [b,h,d,t] so the P@V B-fragments are contiguous loads.
// One thread per (b,h,t,j) pair, j<32.
// ---------------------------------------------------------------------------
__global__ __launch_bounds__(256) void rope_kernel(
    const __bf16* __restrict__ qkv, const int* __restrict__ positions,
    __bf16* __restrict__ qb, __bf16* __restrict__ kb, __bf16* __restrict__ vt)
{
    int gid = blockIdx.x * 256 + threadIdx.x;         // < 2^21
    int j = gid & 31;
    int t = (gid >> 5) & 2047;
    int h = (gid >> 16) & 15;
    int b = gid >> 20;

    float pos = (float)positions[t];
    // inv_freq = 10000^(-j/32) = exp(-j * ln(10000)/32)
    float inv = __expf(-(float)j * 0.28782313662425574f);
    float ang = pos * inv;
    float s, c;
    __sincosf(ang, &s, &c);

    const __bf16* qsrc = qkv;
    const __bf16* ksrc = qkv + 4194304;
    const __bf16* vsrc = qkv + 2 * 4194304;
    size_t tm = (size_t)(b * SEQ + t) * DM + h * DH + j;            // token-major
    size_t hm = ((size_t)(b * NH + h) * SEQ + t) * DH + j;          // head-major

    float q1 = (float)qsrc[tm], q2 = (float)qsrc[tm + 32];
    qb[hm]      = (__bf16)(q1 * c - q2 * s);
    qb[hm + 32] = (__bf16)(q1 * s + q2 * c);

    float k1 = (float)ksrc[tm], k2 = (float)ksrc[tm + 32];
    kb[hm]      = (__bf16)(k1 * c - k2 * s);
    kb[hm + 32] = (__bf16)(k1 * s + k2 * c);

    size_t vth = (size_t)(b * NH + h) * DH;
    vt[(vth + j)      * SEQ + t] = vsrc[tm];
    vt[(vth + j + 32) * SEQ + t] = vsrc[tm + 32];
}

// ---------------------------------------------------------------------------
// Kernel 4: flash attention. One wave per (b, h, 16-query-row tile).
// Scores via 2 WMMAs per 16x16 tile (K=64 head dim); online softmax with
// ALiBi + causal mask computed in registers; P transposed through 1 KB of LDS
// into an A-fragment; P@V via 4 WMMAs per 32-key chunk against V^T.
// With HAVE_ASYNC_LDS: K (4 KB) and V^T (4 KB) tiles are double-buffered in
// LDS via the async DMA path, one chunk ahead of compute.
// ---------------------------------------------------------------------------
__global__ __launch_bounds__(32) void attn_kernel(
    const __bf16* __restrict__ qb, const __bf16* __restrict__ kb,
    const __bf16* __restrict__ vt, __bf16* __restrict__ ob)
{
    __shared__ __bf16 pbuf[16 * 32];
#ifdef HAVE_ASYNC_LDS
    __shared__ __bf16 kbuf[2][32 * 64];     // 2 x 4 KB, rows s (32) x cols d (64)
    __shared__ __bf16 vbuf[2][64 * 32];     // 2 x 4 KB, rows d (64) x cols s (32)
#endif

    const int lane = threadIdx.x;
    const int hs = lane >> 4;
    const int r  = lane & 15;
    int gw = blockIdx.x;                // 0..4095
    int mt = gw & 127;
    int h  = (gw >> 7) & 15;
    int b  = gw >> 11;
    int t0 = mt * 16;

    const __bf16* q   = qb + (size_t)(b * NH + h) * SEQ * DH;
    const __bf16* k   = kb + (size_t)(b * NH + h) * SEQ * DH;
    const __bf16* vtp = vt + (size_t)(b * NH + h) * DH * SEQ;

    // Loop-invariant Q fragments (rows t0..t0+15; d-chunks 0..31, 32..63)
    const __bf16* pq = q + (size_t)(t0 + r) * DH + hs * 16;
    bf16x16 aq0 = load_a_frag(pq);
    bf16x16 aq1 = load_a_frag(pq + 32);

    const float scale = 0.125f;                          // 1/sqrt(64)
    const float slope = exp2f(-0.5f * (float)(h + 1));   // ALiBi slope
    float mrow[8], lrow[8];
    v8f oacc[4] = {};
#pragma unroll
    for (int i = 0; i < 8; ++i) { mrow[i] = -1e30f; lrow[i] = 0.0f; }

#ifdef HAVE_ASYNC_LDS
    // Issue one chunk's K tile (contiguous 4 KB) + V^T tile (64 rows x 64 B):
    // 16 async b128 ops (each lane moves 16 B per op).
    auto issue_tiles = [&](int buf, int s0) {
        const char* gk = (const char*)k + (size_t)s0 * 128;
        char* lk = (char*)&kbuf[buf][0];
#pragma unroll
        for (int it = 0; it < 8; ++it) {
            int off = it * 512 + lane * 16;
            async_cp16(gk + off, lk + off);
        }
        const char* gv = (const char*)vtp + (size_t)s0 * 2;
        char* lv = (char*)&vbuf[buf][0];
#pragma unroll
        for (int it = 0; it < 8; ++it) {
            int row = it * 8 + (lane >> 2);
            int cb  = (lane & 3) * 16;
            async_cp16(gv + (size_t)row * 4096 + cb, lv + row * 64 + cb);
        }
    };
    issue_tiles(0, 0);
#endif

    for (int s0 = 0; s0 < t0 + 16; s0 += 32) {
#ifdef HAVE_ASYNC_LDS
        int buf = (s0 >> 5) & 1;
        if (s0 + 32 < t0 + 16) {
            issue_tiles(buf ^ 1, s0 + 32);              // prefetch next chunk
            asm volatile("s_wait_asynccnt 16" ::: "memory");  // current buf done
        } else {
            asm volatile("s_wait_asynccnt 0" ::: "memory");
        }
        const __bf16* kt  = &kbuf[buf][0];   // + (sub + r)*64 + dc + hs*16
        const __bf16* vbs = &vbuf[buf][0];   // + (j*16 + r)*32 + hs*16
        const int vld = 32;
#else
        __builtin_prefetch(k + (size_t)(s0 + 32 + r) * DH, 0, 1);
        const __bf16* kt  = k + (size_t)s0 * DH;
        const __bf16* vbs = vtp + s0;
        const int vld = SEQ;
#endif
        // ---- scores: two 16x16 subtiles over keys [s0,s0+16) and [s0+16,s0+32)
        v8f c0 = {}, c1 = {};
        {
            const __bf16* pk = kt + (size_t)(r) * DH + hs * 16;
            c0 = wmma_bf16(aq0, load_b_frag(pk), c0);
            c0 = wmma_bf16(aq1, load_b_frag(pk + 32), c0);
        }
        {
            const __bf16* pk = kt + (size_t)(16 + r) * DH + hs * 16;
            c1 = wmma_bf16(aq0, load_b_frag(pk), c1);
            c1 = wmma_bf16(aq1, load_b_frag(pk + 32), c1);
        }

        // ---- ALiBi + causal mask + online softmax (f32 VALU)
#pragma unroll
        for (int i = 0; i < 8; ++i) {
            int t  = t0 + hs * 8 + i;
            int sa = s0 + r;
            int sb = s0 + 16 + r;
            float x0 = c0[i] * scale + slope * (float)(sa - t);
            float x1 = c1[i] * scale + slope * (float)(sb - t);
            if (sa > t) x0 = -1e30f;
            if (sb > t) x1 = -1e30f;
            float mx = fmaxf(x0, x1);
#pragma unroll
            for (int d = 1; d < 16; d <<= 1)
                mx = fmaxf(mx, __shfl_xor(mx, d, 32));
            float nm = fmaxf(mrow[i], mx);
            float f  = __expf(mrow[i] - nm);
            float p0 = __expf(x0 - nm);
            float p1 = __expf(x1 - nm);
            float ps = p0 + p1;
#pragma unroll
            for (int d = 1; d < 16; d <<= 1)
                ps += __shfl_xor(ps, d, 32);
            lrow[i] = lrow[i] * f + ps;
            mrow[i] = nm;
            c0[i] = p0;
            c1[i] = p1;
#pragma unroll
            for (int j = 0; j < 4; ++j) oacc[j][i] *= f;
        }

        // ---- transpose P (16x32) through LDS into A-fragment layout
#pragma unroll
        for (int i = 0; i < 8; ++i) {
            pbuf[(hs * 8 + i) * 32 + r]      = (__bf16)c0[i];
            pbuf[(hs * 8 + i) * 32 + 16 + r] = (__bf16)c1[i];
        }
        __syncthreads();     // single-wave WG: ordering fence only
        bf16x16 ap = load_a_frag(&pbuf[r * 32 + hs * 16]);
        __syncthreads();

        // ---- P @ V : 4 d-strips of 16 against V^T rows (contiguous in s)
#pragma unroll
        for (int j = 0; j < 4; ++j) {
            const __bf16* pv = vbs + (size_t)(j * 16 + r) * vld + hs * 16;
            oacc[j] = wmma_bf16(ap, load_b_frag(pv), oacc[j]);
        }
    }

    // ---- normalize and write token-major bf16 attention output
#pragma unroll
    for (int j = 0; j < 4; ++j) {
#pragma unroll
        for (int i = 0; i < 8; ++i) {
            int t = t0 + hs * 8 + i;
            int d = j * 16 + r;
            ob[(size_t)(b * SEQ + t) * DM + h * DH + d] =
                (__bf16)(oacc[j][i] / lrow[i]);
        }
    }
}

// ---------------------------------------------------------------------------
// Kernel 5: output projection. out = ob @ Wo^T + bo  (f32 out, 32x64 tiles)
// ---------------------------------------------------------------------------
__global__ __launch_bounds__(256) void gemm_out_kernel(
    const __bf16* __restrict__ ab, const __bf16* __restrict__ wo,
    const float* __restrict__ bo, float* __restrict__ out)
{
    const int lane = threadIdx.x & 31;
    const int wave = threadIdx.x >> 5;
    const int hs   = lane >> 4;
    const int r    = lane & 15;
    int gw = blockIdx.x * 8 + wave;
    const int nStrips = DM / 64;      // 16
    int mt = gw / nStrips;            // 0..127
    int ns = gw - mt * nStrips;
    int m0 = mt * 32, n0 = ns * 64;

    v8f acc[2][4] = {};
    const __bf16* pa0 = ab + (size_t)(m0 + r) * DM + hs * 16;
    const __bf16* pa1 = pa0 + (size_t)16 * DM;
    const __bf16* pw  = wo + hs * 16;
    for (int k0 = 0; k0 < DM; k0 += 32) {
        __builtin_prefetch(pa0 + k0 + 256, 0, 1);
        __builtin_prefetch(pa1 + k0 + 256, 0, 1);
        bf16x16 a0 = load_a_frag(pa0 + k0);
        bf16x16 a1 = load_a_frag(pa1 + k0);
#pragma unroll
        for (int j = 0; j < 4; ++j) {
            bf16x16 b = load_b_frag(pw + (size_t)(n0 + j * 16 + r) * DM + k0);
            acc[0][j] = wmma_bf16(a0, b, acc[0][j]);
            acc[1][j] = wmma_bf16(a1, b, acc[1][j]);
        }
    }
#pragma unroll
    for (int j = 0; j < 4; ++j) {
        int nl = n0 + j * 16 + r;
        float bb = bo[nl];
#pragma unroll
        for (int g = 0; g < 2; ++g) {
#pragma unroll
            for (int i = 0; i < 8; ++i) {
                int t = m0 + g * 16 + hs * 8 + i;
                out[(size_t)t * DM + nl] = acc[g][j][i] + bb;
            }
        }
    }
}

// ---------------------------------------------------------------------------
extern "C" void kernel_launch(void* const* d_in, const int* in_sizes, int n_in,
                              void* d_out, int out_size, void* d_ws, size_t ws_size,
                              hipStream_t stream) {
    const float* x   = (const float*)d_in[0];
    const float* Wq  = (const float*)d_in[1];
    const float* bq  = (const float*)d_in[2];
    const float* Wk  = (const float*)d_in[3];
    const float* bk  = (const float*)d_in[4];
    const float* Wv  = (const float*)d_in[5];
    const float* bv  = (const float*)d_in[6];
    const float* Wo  = (const float*)d_in[7];
    const float* bo  = (const float*)d_in[8];
    // d_in[9] = alibi_bias (268 MB) intentionally unread: recomputed in-register
    const int* positions = (const int*)d_in[10];
    float* out = (float*)d_out;

    char* ws = (char*)d_ws;
    __bf16* xb  = (__bf16*)(ws);                      //  8 MB  x bf16
    __bf16* wb  = (__bf16*)(ws + 8388608);            //  8 MB  Wq|Wk|Wv|Wo bf16
    __bf16* qkv = (__bf16*)(ws + 16777216);           // 24 MB  q|k|v token-major
    __bf16* qb  = (__bf16*)(ws + 41943040);           //  8 MB  q head-major (RoPE)
    __bf16* kb  = (__bf16*)(ws + 50331648);           //  8 MB  k head-major (RoPE)
    __bf16* vt  = (__bf16*)(ws + 58720256);           //  8 MB  v^T [b,h,d,t]
    __bf16* ob  = (__bf16*)(ws + 67108864);           //  8 MB  attn out token-major
    const __bf16* wob = wb + 3 * 1048576;             // Wo bf16

    // 1) convert x + weights to bf16 (8M elems / 4 per thread)
    convert_kernel<<<8192, 256, 0, stream>>>(x, Wq, Wk, Wv, Wo, xb, wb);
    // 2) fused QKV GEMM: 128 M-tiles * 48 N-strips = 6144 waves / 8 per block
    gemm_qkv_kernel<<<768, 256, 0, stream>>>(xb, wb, bq, bk, bv, qkv);
    // 3) RoPE + relayout + V transpose (2M threads)
    rope_kernel<<<8192, 256, 0, stream>>>(qkv, positions, qb, kb, vt);
    // 4) flash attention: one wave per (b,h,16-row tile) = 2*16*128 blocks
    attn_kernel<<<4096, 32, 0, stream>>>(qb, kb, vt, ob);
    // 5) output projection: 128 * 16 = 2048 waves / 8 per block
    gemm_out_kernel<<<256, 256, 0, stream>>>(ob, wob, bo, out);
}
